// SentenceNet_55070070670236
// MI455X (gfx1250) — compile-verified, matrix operand order checked by reading, not run on hardware
//
#include <hip/hip_runtime.h>
#include <hip/hip_bf16.h>
#include <math.h>

typedef __bf16 bf16;
typedef __attribute__((ext_vector_type(16))) __bf16 v16bf;
typedef __attribute__((ext_vector_type(8)))  float  v8f;

#define HC    5120      // H*C
#define NSENT 4096
#define NWORD 65536
#define NSEC  512
#define KWPAD 320       // K=300 padded to multiple of 32

// ---------------------------------------------------------------- utilities
__global__ void k_f2b(const float* __restrict__ in, bf16* __restrict__ out, long long n) {
    long long i = (long long)blockIdx.x * blockDim.x + threadIdx.x;
    if (i < n) out[i] = (bf16)in[i];
}

// rows x K (f32) -> rows x Kpad (bf16), zero-padded tail
__global__ void k_f2b_pad(const float* __restrict__ in, bf16* __restrict__ out,
                          long long rows, int K, int Kpad) {
    long long i = (long long)blockIdx.x * blockDim.x + threadIdx.x;
    if (i >= rows * Kpad) return;
    long long r = i / Kpad; int k = (int)(i % Kpad);
    out[i] = (k < K) ? (bf16)in[r * K + k] : (bf16)0.0f;
}

// W[K][Ncols] (f32) -> Wt[Ncols][Kpad] (bf16), zero-padded tail
__global__ void k_transb(const float* __restrict__ in, bf16* __restrict__ out,
                         int K, int Kpad, int Ncols) {
    long long i = (long long)blockIdx.x * blockDim.x + threadIdx.x;
    long long total = (long long)Kpad * Ncols;
    if (i >= total) return;
    int n = (int)(i / Kpad), k = (int)(i % Kpad);
    out[i] = (k < K) ? (bf16)in[(long long)k * Ncols + n] : (bf16)0.0f;
}

__global__ void k_fill(float* __restrict__ p, float v, long long n) {
    long long i = (long long)blockIdx.x * blockDim.x + threadIdx.x;
    if (i < n) p[i] = v;
}

__global__ void k_concat(const bf16* __restrict__ A, const bf16* __restrict__ B,
                         bf16* __restrict__ out, long long total) {
    long long i = (long long)blockIdx.x * blockDim.x + threadIdx.x;
    if (i >= total) return;
    long long m = i / (2 * HC); int c = (int)(i % (2 * HC));
    out[i] = (c < HC) ? A[m * HC + c] : B[m * HC + (c - HC)];
}

// agg += bias (per column), ELU in place, plus bf16 copy
__global__ void k_elubias(float* __restrict__ agg, bf16* __restrict__ outB,
                          const float* __restrict__ bias, long long total) {
    long long i = (long long)blockIdx.x * blockDim.x + threadIdx.x;
    if (i >= total) return;
    float v = agg[i] + bias[(int)(i % HC)];
    v = (v > 0.0f) ? v : expm1f(v);
    agg[i] = v;
    outB[i] = (bf16)v;
}

// ------------------------------------------------------------ WMMA GEMM
// C[M][N] = A[M][K] @ Bt[N][K]^T, bf16 in, f32 accumulate.
// 256 threads = 8 waves (2 M x 4 N), block tile 128x128, wave tile 64x32,
// K-step 32. Requires M%128==0, N%128==0, K%32==0 (K pre-padded).
__device__ __forceinline__ v16bf frag16(const bf16* __restrict__ t, int lane) {
    // CDNA5 16-bit A/B layout: lane<16 -> row=lane, K chunks {0..7},{16..23};
    // lane>=16 -> row=lane-16, K chunks {8..15},{24..31}. Two 16B LDS loads.
    int r = lane & 15, kh = lane >> 4;
    const bf16* p = t + r * 40 + kh * 8;
    union { v16bf v; float4 q[2]; } u;
    u.q[0] = *(const float4*)(p);
    u.q[1] = *(const float4*)(p + 16);
    return u.v;
}

// epi: 0 = (+bias) store, 1 = ReLU, 2 = sigmoid gate (auxA/auxB), 3 = +res
__global__ __launch_bounds__(256) void k_gemm(
    const bf16* __restrict__ A, const bf16* __restrict__ Bt,
    const float* __restrict__ bias,
    float* __restrict__ outF, bf16* __restrict__ outB,
    const float* __restrict__ auxA, const float* __restrict__ auxB,
    const float* __restrict__ res,
    int M, int N, int K, int epi)
{
    __shared__ __align__(16) bf16 lA[128][40];
    __shared__ __align__(16) bf16 lB[128][40];

    int tid  = threadIdx.x;
    int lane = tid & 31, wid = tid >> 5;
    int wm = (wid & 1) * 64, wn = (wid >> 1) * 32;
    int m0 = blockIdx.x * 128, n0 = blockIdx.y * 128;

    // straight-line staging assignment: thread -> rows (r, r+64), one 8-elem col
    int rs  = tid >> 2;              // 0..63
    int cs  = (tid & 3) * 8;         // 0,8,16,24
    const bf16* gA0 = A  + (long long)(m0 + rs)      * K + cs;
    const bf16* gA1 = A  + (long long)(m0 + rs + 64) * K + cs;
    const bf16* gB0 = Bt + (long long)(n0 + rs)      * K + cs;
    const bf16* gB1 = Bt + (long long)(n0 + rs + 64) * K + cs;

    v8f acc[4][2] = {};

    for (int kt = 0; kt < K; kt += 32) {
        *(float4*)&lA[rs][cs]      = *(const float4*)(gA0 + kt);
        *(float4*)&lA[rs + 64][cs] = *(const float4*)(gA1 + kt);
        *(float4*)&lB[rs][cs]      = *(const float4*)(gB0 + kt);
        *(float4*)&lB[rs + 64][cs] = *(const float4*)(gB1 + kt);
        // speculative prefetch of the next K-tile (safe past end: DEV scope)
        __builtin_prefetch(gA0 + kt + 32, 0, 1);
        __builtin_prefetch(gA1 + kt + 32, 0, 1);
        __builtin_prefetch(gB0 + kt + 32, 0, 1);
        __builtin_prefetch(gB1 + kt + 32, 0, 1);
        __syncthreads();

        v16bf af[4], bfr[2];
        #pragma unroll
        for (int i = 0; i < 4; ++i) af[i]  = frag16(&lA[wm + i * 16][0], lane);
        #pragma unroll
        for (int j = 0; j < 2; ++j) bfr[j] = frag16(&lB[wn + j * 16][0], lane);
        #pragma unroll
        for (int i = 0; i < 4; ++i)
            #pragma unroll
            for (int j = 0; j < 2; ++j)
                acc[i][j] = __builtin_amdgcn_wmma_f32_16x16x32_bf16(
                    false, af[i], false, bfr[j], (short)0, acc[i][j], false, false);
        __syncthreads();
    }

    // C layout: VGPR r -> (M = mh*8 + r, N = lane&15)
    int nl = lane & 15, mh = lane >> 4;
    #pragma unroll
    for (int i = 0; i < 4; ++i)
        #pragma unroll
        for (int j = 0; j < 2; ++j) {
            int ncol = n0 + wn + j * 16 + nl;
            float bv = bias ? bias[ncol] : 0.0f;
            #pragma unroll
            for (int r = 0; r < 8; ++r) {
                int mrow = m0 + wm + i * 16 + mh * 8 + r;
                long long idx = (long long)mrow * N + ncol;
                float v = acc[i][j][r] + bv;
                if (epi == 1) {
                    v = v > 0.0f ? v : 0.0f;
                } else if (epi == 2) {
                    float z = 1.0f / (1.0f + __expf(-v));
                    v = z * auxA[idx] + (1.0f - z) * auxB[idx];
                } else if (epi == 3) {
                    v = v + res[idx];
                }
                if (outF) outF[idx] = v;
                if (outB) outB[idx] = (bf16)v;
            }
        }
}

// ----------------------------------------------------- attention reductions
// out[row][h] = sum_c P[row][h*640+c] * a[h*640+c]; one wave per head
__global__ __launch_bounds__(256) void k_headdot(
    const bf16* __restrict__ P, const float* __restrict__ a,
    float* __restrict__ out, int rows)
{
    int row = blockIdx.x;
    if (row >= rows) return;
    int wid = threadIdx.x >> 5, lane = threadIdx.x & 31;
    const bf16*  p  = P + (long long)row * HC + wid * 640;
    const float* av = a + wid * 640;
    float s = 0.0f;
    for (int c = lane; c < 640; c += 32) s += (float)p[c] * av[c];
    #pragma unroll
    for (int off = 16; off > 0; off >>= 1) s += __shfl_down(s, off, 32);
    if (lane == 0) out[row * 8 + wid] = s;
}

__device__ __forceinline__ void atomicMaxF(float* addr, float v) {
    if (v >= 0.0f) atomicMax((int*)addr, __float_as_int(v));
    else           atomicMin((unsigned int*)addr, (unsigned int)__float_as_int(v));
}

__global__ void k_elogits(const int* __restrict__ edges, int E,
                          const float* __restrict__ als, const float* __restrict__ ald,
                          float* __restrict__ ebuf, float* __restrict__ mmax)
{
    long long i = (long long)blockIdx.x * blockDim.x + threadIdx.x;
    if (i >= (long long)E * 8) return;
    int e = (int)(i >> 3), h = (int)(i & 7);
    int s = edges[e], d = edges[E + e];
    float v = als[s * 8 + h] + ald[d * 8 + h];
    v = (v > 0.0f) ? v : 0.2f * v;          // leaky_relu 0.2
    ebuf[i] = v;
    atomicMaxF(&mmax[d * 8 + h], v);
}

__global__ void k_eexp(const int* __restrict__ edges, int E,
                       float* __restrict__ ebuf, const float* __restrict__ mmax,
                       float* __restrict__ den)
{
    long long i = (long long)blockIdx.x * blockDim.x + threadIdx.x;
    if (i >= (long long)E * 8) return;
    int e = (int)(i >> 3), h = (int)(i & 7);
    int d = edges[E + e];
    float ex = __expf(ebuf[i] - mmax[d * 8 + h]);
    ebuf[i] = ex;
    atomicAdd(&den[d * 8 + h], ex);
}

// out[dst][h*640+c] += hs[src][h*640+c] * alpha ; thread = (edge, head, 8-chan chunk)
__global__ void k_eagg(const int* __restrict__ edges, int E,
                       const float* __restrict__ ebuf, const float* __restrict__ den,
                       const bf16* __restrict__ hsrc, float* __restrict__ out)
{
    long long i = (long long)blockIdx.x * blockDim.x + threadIdx.x;
    if (i >= (long long)E * 640) return;
    int c8 = (int)(i % 80);
    int h  = (int)((i / 80) % 8);
    int e  = (int)(i / 640);
    int s = edges[e], d = edges[E + e];
    float alpha = ebuf[e * 8 + h] / (den[d * 8 + h] + 1e-16f);
    const bf16* src = hsrc + (long long)s * HC + h * 640 + c8 * 8;
    float*      dst = out  + (long long)d * HC + h * 640 + c8 * 8;
    #pragma unroll
    for (int j = 0; j < 8; ++j) atomicAdd(dst + j, (float)src[j] * alpha);
}

// ---------------------------------------------------------------- launcher
extern "C" void kernel_launch(void* const* d_in, const int* in_sizes, int n_in,
                              void* d_out, int out_size, void* d_ws, size_t ws_size,
                              hipStream_t stream) {
    const float* Hs   = (const float*)d_in[0];
    const float* Hw   = (const float*)d_in[1];
    const float* HSd  = (const float*)d_in[2];
    const int*   e_w  = (const int*)d_in[3];   // 2 x 65536
    const int*   e_s  = (const int*)d_in[4];   // 2 x 32768
    const int*   e_S  = (const int*)d_in[5];   // 2 x 4096
    const float* Ww_src=(const float*)d_in[6];
    const float* Ww_dst=(const float*)d_in[7];
    const float* aw_src=(const float*)d_in[8];
    const float* aw_dst=(const float*)d_in[9];
    const float* bw   =(const float*)d_in[10];
    const float* Wsm  =(const float*)d_in[11];
    const float* as_src=(const float*)d_in[12];
    const float* as_dst=(const float*)d_in[13];
    const float* bs   =(const float*)d_in[14];
    const float* WS_src=(const float*)d_in[15];
    const float* WS_dst=(const float*)d_in[16];
    const float* aS_src=(const float*)d_in[17];
    const float* aS_dst=(const float*)d_in[18];
    const float* bS   =(const float*)d_in[19];
    const float* Wf1  =(const float*)d_in[20];
    const float* bf1  =(const float*)d_in[21];
    const float* Wf2  =(const float*)d_in[22];
    const float* bf2  =(const float*)d_in[23];
    const float* W1   =(const float*)d_in[24];
    const float* b1   =(const float*)d_in[25];
    const float* W2   =(const float*)d_in[26];
    const float* b2   =(const float*)d_in[27];

    const int EW = 65536, ES = 32768, ESS = 4096;

    // bump allocator over d_ws
    char* base = (char*)d_ws;
    size_t off = 0;
    auto alloc = [&](size_t bytes) -> void* {
        void* p = base + off;
        off += (bytes + 255) & ~(size_t)255;
        return p;
    };

    bf16* Hw_b  = (bf16*)alloc((size_t)NWORD * KWPAD * 2);   // padded K
    bf16* Hs_b  = (bf16*)alloc((size_t)NSENT * 640 * 2);
    bf16* HS_b  = (bf16*)alloc((size_t)NSEC  * 512 * 2);
    bf16* WwsT  = (bf16*)alloc((size_t)HC * KWPAD * 2);      // padded K
    bf16* WwdT  = (bf16*)alloc((size_t)HC * 640 * 2);
    bf16* WsT   = (bf16*)alloc((size_t)HC * 640 * 2);
    bf16* WSsT  = (bf16*)alloc((size_t)HC * 512 * 2);
    bf16* WSdT  = (bf16*)alloc((size_t)HC * 640 * 2);
    bf16* Wf1T  = (bf16*)alloc((size_t)HC * 2 * HC * 2);
    bf16* Wf2T  = (bf16*)alloc((size_t)HC * 2 * HC * 2);
    bf16* W1T   = (bf16*)alloc((size_t)640 * HC * 2);
    bf16* W2T   = (bf16*)alloc((size_t)640 * 640 * 2);
    bf16* Qw    = (bf16*)alloc((size_t)NSENT * HC * 2);
    bf16* Ps    = (bf16*)alloc((size_t)NSENT * HC * 2);
    bf16* PSs   = (bf16*)alloc((size_t)NSEC  * HC * 2);
    bf16* QSd   = (bf16*)alloc((size_t)NSENT * HC * 2);
    float* alw_s= (float*)alloc((size_t)NWORD * 8 * 4);
    float* alw_d= (float*)alloc((size_t)NSENT * 8 * 4);
    float* als_s= (float*)alloc((size_t)NSENT * 8 * 4);
    float* als_d= (float*)alloc((size_t)NSENT * 8 * 4);
    float* alS_s= (float*)alloc((size_t)NSEC  * 8 * 4);
    float* alS_d= (float*)alloc((size_t)NSENT * 8 * 4);
    float* m_w  = (float*)alloc((size_t)NSENT * 8 * 4);
    float* d_w  = (float*)alloc((size_t)NSENT * 8 * 4);
    float* m_s  = (float*)alloc((size_t)NSENT * 8 * 4);
    float* d_s  = (float*)alloc((size_t)NSENT * 8 * 4);
    float* m_S  = (float*)alloc((size_t)NSENT * 8 * 4);
    float* d_S  = (float*)alloc((size_t)NSENT * 8 * 4);
    float* ebw  = (float*)alloc((size_t)EW  * 8 * 4);
    float* ebs  = (float*)alloc((size_t)ES  * 8 * 4);
    float* ebS  = (float*)alloc((size_t)ESS * 8 * 4);
    float* aggw = (float*)alloc((size_t)NSENT * HC * 4);
    float* aggs = (float*)alloc((size_t)NSENT * HC * 4);
    float* aggS = (float*)alloc((size_t)NSENT * HC * 4);
    bf16* Uw_b  = (bf16*)alloc((size_t)NSENT * HC * 2);
    bf16* Us_b  = (bf16*)alloc((size_t)NSENT * HC * 2);
    bf16* US_b  = (bf16*)alloc((size_t)NSENT * HC * 2);
    bf16* Pw    = (bf16*)alloc((size_t)NWORD * HC * 2);   // 671 MB, dead after word GAT
    // fusion-stage temporaries carved out of the (then dead) Pw region
    char* fb = (char*)Pw;
    bf16*  Cat1 = (bf16*)(fb);                         fb += (size_t)NSENT * 2 * HC * 2;
    bf16*  Cat2 = (bf16*)(fb);                         fb += (size_t)NSENT * 2 * HC * 2;
    float* U1f  = (float*)(fb);                        fb += (size_t)NSENT * HC * 4;
    float* U2f  = (float*)(fb);                        fb += (size_t)NSENT * HC * 4;
    bf16*  U1b  = (bf16*)(fb);                         fb += (size_t)NSENT * HC * 2;
    bf16*  U2b  = (bf16*)(fb);                         fb += (size_t)NSENT * HC * 2;
    bf16*  Hid  = (bf16*)(fb);                         fb += (size_t)NSENT * 640 * 2;

    auto blocks = [](long long n) { return dim3((unsigned)((n + 255) / 256)); };

    // ---- convert activations + transpose-convert weights to bf16 (K padded to %32)
    k_f2b_pad<<<blocks((long long)NWORD*KWPAD), 256, 0, stream>>>(Hw, Hw_b, NWORD, 300, KWPAD);
    k_f2b<<<blocks((long long)NSENT*640), 256, 0, stream>>>(Hs, Hs_b, (long long)NSENT*640);
    k_f2b<<<blocks((long long)NSEC*512),  256, 0, stream>>>(HSd, HS_b, (long long)NSEC*512);
    k_transb<<<blocks((long long)KWPAD*HC), 256, 0, stream>>>(Ww_src, WwsT, 300, KWPAD, HC);
    k_transb<<<blocks((long long)640*HC), 256, 0, stream>>>(Ww_dst, WwdT, 640, 640, HC);
    k_transb<<<blocks((long long)640*HC), 256, 0, stream>>>(Wsm,    WsT,  640, 640, HC);
    k_transb<<<blocks((long long)512*HC), 256, 0, stream>>>(WS_src, WSsT, 512, 512, HC);
    k_transb<<<blocks((long long)640*HC), 256, 0, stream>>>(WS_dst, WSdT, 640, 640, HC);
    k_transb<<<blocks((long long)2*HC*HC),256, 0, stream>>>(Wf1, Wf1T, 2*HC, 2*HC, HC);
    k_transb<<<blocks((long long)2*HC*HC),256, 0, stream>>>(Wf2, Wf2T, 2*HC, 2*HC, HC);
    k_transb<<<blocks((long long)HC*640), 256, 0, stream>>>(W1, W1T, HC, HC, 640);
    k_transb<<<blocks((long long)640*640),256, 0, stream>>>(W2, W2T, 640, 640, 640);

    // ---- projections (WMMA GEMMs, bf16 out)
    k_gemm<<<dim3(NWORD/128, HC/128), 256, 0, stream>>>(Hw_b, WwsT, nullptr, nullptr, Pw,
        nullptr, nullptr, nullptr, NWORD, HC, KWPAD, 0);
    k_gemm<<<dim3(NSENT/128, HC/128), 256, 0, stream>>>(Hs_b, WwdT, nullptr, nullptr, Qw,
        nullptr, nullptr, nullptr, NSENT, HC, 640, 0);
    k_gemm<<<dim3(NSENT/128, HC/128), 256, 0, stream>>>(Hs_b, WsT, nullptr, nullptr, Ps,
        nullptr, nullptr, nullptr, NSENT, HC, 640, 0);
    k_gemm<<<dim3(NSEC/128, HC/128), 256, 0, stream>>>(HS_b, WSsT, nullptr, nullptr, PSs,
        nullptr, nullptr, nullptr, NSEC, HC, 512, 0);
    k_gemm<<<dim3(NSENT/128, HC/128), 256, 0, stream>>>(Hs_b, WSdT, nullptr, nullptr, QSd,
        nullptr, nullptr, nullptr, NSENT, HC, 640, 0);

    // ---- attention coefficients
    k_headdot<<<NWORD, 256, 0, stream>>>(Pw,  aw_src, alw_s, NWORD);
    k_headdot<<<NSENT, 256, 0, stream>>>(Qw,  aw_dst, alw_d, NSENT);
    k_headdot<<<NSENT, 256, 0, stream>>>(Ps,  as_src, als_s, NSENT);
    k_headdot<<<NSENT, 256, 0, stream>>>(Ps,  as_dst, als_d, NSENT);
    k_headdot<<<NSEC,  256, 0, stream>>>(PSs, aS_src, alS_s, NSEC);
    k_headdot<<<NSENT, 256, 0, stream>>>(QSd, aS_dst, alS_d, NSENT);

    // ---- init segment buffers
    const long long n8 = (long long)NSENT * 8;
    k_fill<<<blocks(n8), 256, 0, stream>>>(m_w, -INFINITY, n8);
    k_fill<<<blocks(n8), 256, 0, stream>>>(m_s, -INFINITY, n8);
    k_fill<<<blocks(n8), 256, 0, stream>>>(m_S, -INFINITY, n8);
    k_fill<<<blocks(n8), 256, 0, stream>>>(d_w, 0.0f, n8);
    k_fill<<<blocks(n8), 256, 0, stream>>>(d_s, 0.0f, n8);
    k_fill<<<blocks(n8), 256, 0, stream>>>(d_S, 0.0f, n8);
    const long long nAgg = (long long)NSENT * HC;
    k_fill<<<blocks(nAgg), 256, 0, stream>>>(aggw, 0.0f, nAgg);
    k_fill<<<blocks(nAgg), 256, 0, stream>>>(aggs, 0.0f, nAgg);
    k_fill<<<blocks(nAgg), 256, 0, stream>>>(aggS, 0.0f, nAgg);

    // ---- edge softmax + aggregation (three GATs)
    k_elogits<<<blocks((long long)EW*8),  256, 0, stream>>>(e_w, EW,  alw_s, alw_d, ebw, m_w);
    k_elogits<<<blocks((long long)ES*8),  256, 0, stream>>>(e_s, ES,  als_s, als_d, ebs, m_s);
    k_elogits<<<blocks((long long)ESS*8), 256, 0, stream>>>(e_S, ESS, alS_s, alS_d, ebS, m_S);
    k_eexp<<<blocks((long long)EW*8),  256, 0, stream>>>(e_w, EW,  ebw, m_w, d_w);
    k_eexp<<<blocks((long long)ES*8),  256, 0, stream>>>(e_s, ES,  ebs, m_s, d_s);
    k_eexp<<<blocks((long long)ESS*8), 256, 0, stream>>>(e_S, ESS, ebS, m_S, d_S);
    k_eagg<<<blocks((long long)EW*640),  256, 0, stream>>>(e_w, EW,  ebw, d_w, Pw,  aggw);
    k_eagg<<<blocks((long long)ES*640),  256, 0, stream>>>(e_s, ES,  ebs, d_s, Ps,  aggs);
    k_eagg<<<blocks((long long)ESS*640), 256, 0, stream>>>(e_S, ESS, ebS, d_S, PSs, aggS);

    // ---- bias + ELU -> Uw/Us/US (f32 in place) + bf16 copies
    k_elubias<<<blocks(nAgg), 256, 0, stream>>>(aggw, Uw_b, bw, nAgg);
    k_elubias<<<blocks(nAgg), 256, 0, stream>>>(aggs, Us_b, bs, nAgg);
    k_elubias<<<blocks(nAgg), 256, 0, stream>>>(aggS, US_b, bS, nAgg);

    // ---- fusion 1: z = sigmoid([Uw;Us]@Wf1 + bf1); U1 = z*Uw + (1-z)*Us
    k_concat<<<blocks((long long)NSENT*2*HC), 256, 0, stream>>>(Uw_b, Us_b, Cat1,
                                                               (long long)NSENT*2*HC);
    k_gemm<<<dim3(NSENT/128, HC/128), 256, 0, stream>>>(Cat1, Wf1T, bf1, U1f, U1b,
        aggw, aggs, nullptr, NSENT, HC, 2*HC, 2);

    // ---- fusion 2: U2 = gate(U1, US)
    k_concat<<<blocks((long long)NSENT*2*HC), 256, 0, stream>>>(U1b, US_b, Cat2,
                                                               (long long)NSENT*2*HC);
    k_gemm<<<dim3(NSENT/128, HC/128), 256, 0, stream>>>(Cat2, Wf2T, bf2, U2f, U2b,
        U1f, aggS, nullptr, NSENT, HC, 2*HC, 2);

    // ---- FFN: Hid = relu(U2 @ W1 + b1);  out = Hs + Hid @ W2 + b2
    k_gemm<<<dim3(NSENT/128, 640/128), 256, 0, stream>>>(U2b, W1T, b1, nullptr, Hid,
        nullptr, nullptr, nullptr, NSENT, 640, HC, 1);
    k_gemm<<<dim3(NSENT/128, 640/128), 256, 0, stream>>>(Hid, W2T, b2, (float*)d_out, nullptr,
        nullptr, nullptr, Hs, NSENT, 640, 640, 3);

    (void)in_sizes; (void)n_in; (void)out_size; (void)ws_size;
}
